// Sequence_33474975105038
// MI455X (gfx1250) — matrix operand order repaired and checked by reading, not compiled, not run on hardware
//
#include <hip/hip_runtime.h>

// ---------------------------------------------------------------------------
// 2-layer LSTM (T=2048, B=4096, H=51) for gfx1250.
// Persistent blocks: 64 blocks x 128 threads (4 wave32), each block owns 64
// batch rows for the whole sequence. Recurrent matmuls use
// v_wmma_f32_16x16x32_f16 with f16 weights/activations pre-staged in LDS in
// native fragment layout; f32 accumulation; state (h,c) lives in LDS.
// ---------------------------------------------------------------------------

typedef __attribute__((ext_vector_type(16))) _Float16 v16h;
typedef __attribute__((ext_vector_type(8)))  _Float16 v8h;
typedef __attribute__((ext_vector_type(8)))  float    v8f;

#define TSTEPS   2048
#define BATCH    4096
#define HID      51          // hidden size
#define GATES    204         // 4*HID
#define GP       208         // gates padded to 13 N-tiles of 16
#define NT       13          // N tiles
#define KP       64          // K (hidden) padded: 2 k-chunks of 32
#define ROWS     64          // batch rows per block
#define NTHREADS 128         // 4 wave32
#define CSTRIDE  52          // c-state row stride (f32)

// ---- dynamic LDS layout (all offsets 16B aligned) -------------------------
constexpr int OFF_W1   = 0;                           // [2][13][32][16] f16
constexpr int OFF_W2   = OFF_W1 + 2 * NT * 512 * 2;   // 26624: [4][13][32][16] f16
constexpr int OFF_G    = OFF_W2 + 4 * NT * 512 * 2;   // 79872: [64][208] f32
constexpr int OFF_H1   = OFF_G + ROWS * GP * 4;       // 133120: [64][64] f16
constexpr int OFF_H2   = OFF_H1 + ROWS * KP * 2;      // 141312: [64][64] f16
constexpr int OFF_C1   = OFF_H2 + ROWS * KP * 2;      // 149504: [64][52] f32
constexpr int OFF_C2   = OFF_C1 + ROWS * CSTRIDE * 4; // 162816: [64][52] f32
constexpr int OFF_X    = OFF_C2 + ROWS * CSTRIDE * 4; // 176128: [64][2] f32
constexpr int OFF_B1   = OFF_X + ROWS * 2 * 4;        // 176640: [208] f32
constexpr int OFF_B2   = OFF_B1 + GP * 4;             // 177472: [208] f32
constexpr int OFF_WI1  = OFF_B2 + GP * 4;             // 178304: [208][2] f32
constexpr int OFF_WLIN = OFF_WI1 + GP * 2 * 4;        // 179968: [64] f32
constexpr int OFF_BLIN = OFF_WLIN + 64 * 4;           // 180224: [4] f32
constexpr int OFF_OUT  = OFF_BLIN + 16;               // 180240: [64][16] f32
constexpr int SMEM_BYTES = OFF_OUT + ROWS * 16 * 4;   // 184336 B (< 320 KB/WGP)

// ---- fast transcendental helpers (v_exp_f32 / v_rcp_f32) ------------------
__device__ __forceinline__ float fsigmoid(float x) {
    return __builtin_amdgcn_rcpf(1.0f + __builtin_amdgcn_exp2f(-1.44269504f * x));
}
__device__ __forceinline__ float ftanh(float x) {
    x = fminf(10.0f, fmaxf(-10.0f, x));
    float e = __builtin_amdgcn_exp2f(2.88539008f * x);   // exp(2x)
    return (e - 1.0f) * __builtin_amdgcn_rcpf(e + 1.0f);
}

// ---- WMMA wrappers --------------------------------------------------------
__device__ __forceinline__ v8f wmma_f16(v16h a, v16h b, v8f c) {
    // D = A(16x32 f16) * B(32x16 f16) + C(16x16 f32)
    return __builtin_amdgcn_wmma_f32_16x16x32_f16(
        /*neg_a=*/false, a, /*neg_b=*/false, b,
        /*c_mod=*/(short)0, c, /*reuse_a=*/false, /*reuse_b=*/false);
}

// A-fragment from h-state LDS ([row][64] f16, row-major).
// ISA 16-bit A layout: lanes 0-15 -> K {0..7,16..23}+8*hi for lanes 16-31.
__device__ __forceinline__ v16h load_afrag(const _Float16* hbuf, int row, int kbase, int hi) {
    const _Float16* p = hbuf + row * KP + kbase + (hi << 3);
    union { v16h v; v8h h[2]; } u;
    u.h[0] = *(const v8h*)(p);        // K = kbase + 8*hi + 0..7
    u.h[1] = *(const v8h*)(p + 16);   // K = kbase + 8*hi + 16..23
    return u.v;
}

// B-fragment from pre-permuted weight LDS: [kc][nt][lane][16] f16, contiguous.
__device__ __forceinline__ v16h load_bfrag(const _Float16* wf, int kc, int nt, int lane) {
    const _Float16* p = wf + (((kc * NT + nt) * 32 + lane) << 4);
    union { v16h v; v8h h[2]; } u;
    u.h[0] = *(const v8h*)(p);
    u.h[1] = *(const v8h*)(p + 8);
    return u.v;
}

// D(f32 16x16) -> gates LDS [64][GP]: VGPR r = row (r or r+8), lane&15 = col.
__device__ __forceinline__ void store_cfrag(float* gates, int row0, int nt, int lane, v8f c) {
    int hi = lane >> 4;
    int n  = nt * 16 + (lane & 15);
    float* p = gates + (row0 + (hi << 3)) * GP + n;
#pragma unroll
    for (int r = 0; r < 8; ++r) p[r * GP] = c[r];
}

// elementwise LSTM cell update: gates [64][GP] -> c (f32 LDS), h (f16 LDS)
__device__ __forceinline__ void ew_update(const float* gates, float* cbuf,
                                          _Float16* hbuf, int tid) {
    for (int idx = tid; idx < ROWS * HID; idx += NTHREADS) {
        int row = idx / HID;
        int j   = idx - row * HID;
        const float* g = gates + row * GP;
        float ig = fsigmoid(g[j]);
        float fg = fsigmoid(g[HID + j]);
        float gg = ftanh(g[2 * HID + j]);
        float og = fsigmoid(g[3 * HID + j]);
        float cn = fg * cbuf[row * CSTRIDE + j] + ig * gg;
        cbuf[row * CSTRIDE + j] = cn;
        hbuf[row * KP + j] = (_Float16)(og * ftanh(cn));
    }
}

__global__ __launch_bounds__(NTHREADS, 1)
void lstm2_wmma_kernel(const float* __restrict__ input,
                       const float* __restrict__ Wih1, const float* __restrict__ Whh1,
                       const float* __restrict__ bih1, const float* __restrict__ bhh1,
                       const float* __restrict__ Wih2, const float* __restrict__ Whh2,
                       const float* __restrict__ bih2, const float* __restrict__ bhh2,
                       const float* __restrict__ Wlin, const float* __restrict__ blin,
                       float* __restrict__ out) {
    extern __shared__ char smem[];
    _Float16* w1f   = (_Float16*)(smem + OFF_W1);
    _Float16* w2f   = (_Float16*)(smem + OFF_W2);
    float*    gates = (float*)(smem + OFF_G);
    _Float16* h1    = (_Float16*)(smem + OFF_H1);
    _Float16* h2    = (_Float16*)(smem + OFF_H2);
    float*    c1    = (float*)(smem + OFF_C1);
    float*    c2    = (float*)(smem + OFF_C2);
    float*    xb    = (float*)(smem + OFF_X);
    float*    bias1 = (float*)(smem + OFF_B1);
    float*    bias2 = (float*)(smem + OFF_B2);
    float*    wi1   = (float*)(smem + OFF_WI1);
    float*    wlin  = (float*)(smem + OFF_WLIN);
    float*    blinS = (float*)(smem + OFF_BLIN);
    float*    obuf  = (float*)(smem + OFF_OUT);

    const int tid   = threadIdx.x;
    const int lane  = tid & 31;
    const int wave  = tid >> 5;
    const int gbase = blockIdx.x * ROWS;

    // ------------- one-time prep: stage weights as WMMA B-fragments --------
    // layer-1 recurrent weights: B[k][n] = Whh1[n][k], K padded to 64.
    for (int idx = tid; idx < 2 * NT * 512; idx += NTHREADS) {
        int e  = idx & 15;
        int ln = (idx >> 4) & 31;
        int r  = idx >> 9;                // kc*NT + nt
        int nt = r % NT, kc = r / NT;
        int n  = nt * 16 + (ln & 15);
        int k  = kc * 32 + (e & 7) + ((e >> 3) << 4) + ((ln >> 4) << 3);
        float v = (n < GATES && k < HID) ? Whh1[n * HID + k] : 0.0f;
        w1f[idx] = (_Float16)v;
    }
    // layer-2 stacked weights: K 0..63 = Wih2 (input h1), K 64..127 = Whh2.
    for (int idx = tid; idx < 4 * NT * 512; idx += NTHREADS) {
        int e  = idx & 15;
        int ln = (idx >> 4) & 31;
        int r  = idx >> 9;
        int nt = r % NT, kc = r / NT;
        int n  = nt * 16 + (ln & 15);
        int k  = kc * 32 + (e & 7) + ((e >> 3) << 4) + ((ln >> 4) << 3);   // 0..127
        float v = 0.0f;
        if (n < GATES) {
            if (k < KP) { if (k < HID)       v = Wih2[n * HID + k]; }
            else        { int kk = k - KP; if (kk < HID) v = Whh2[n * HID + kk]; }
        }
        w2f[idx] = (_Float16)v;
    }
    for (int n = tid; n < GP; n += NTHREADS) {
        bias1[n]       = (n < GATES) ? bih1[n] + bhh1[n] : 0.0f;
        bias2[n]       = (n < GATES) ? bih2[n] + bhh2[n] : 0.0f;
        wi1[2 * n]     = (n < GATES) ? Wih1[2 * n]     : 0.0f;
        wi1[2 * n + 1] = (n < GATES) ? Wih1[2 * n + 1] : 0.0f;
    }
    for (int j = tid; j < KP; j += NTHREADS) wlin[j] = (j < HID) ? Wlin[j] : 0.0f;
    if (tid == 0) blinS[0] = blin[0];
    for (int i = tid; i < ROWS * KP; i += NTHREADS) { h1[i] = (_Float16)0.0f; h2[i] = (_Float16)0.0f; }
    for (int i = tid; i < ROWS * CSTRIDE; i += NTHREADS) { c1[i] = 0.0f; c2[i] = 0.0f; }
    __syncthreads();

    const int hi   = lane >> 4;
    const int m    = lane & 15;
    const int row0 = wave * 16;          // this wave's 16-row tile

    // ------------- time recurrence -----------------------------------------
    for (int t = 0; t < TSTEPS; ++t) {
        // stage x_t (64 rows x 2 feats = 128 f32, fully coalesced)
        const float* xg = input + ((size_t)t * BATCH + gbase) * 2;
        xb[tid] = xg[tid];
        if (t + 1 < TSTEPS)
            __builtin_prefetch(xg + (size_t)BATCH * 2 + tid, 0, 1);
        __syncthreads();

        // ---- layer 1: gates = h1 @ Whh1^T + (b + x @ Wih1^T) --------------
        {
            v16h a0 = load_afrag(h1, row0 + m, 0,  hi);
            v16h a1 = load_afrag(h1, row0 + m, 32, hi);
#pragma unroll
            for (int nt = 0; nt < NT; ++nt) {
                int n = nt * 16 + m;
                float bz = bias1[n];
                float w0 = wi1[2 * n], w1 = wi1[2 * n + 1];
                v8f c;
#pragma unroll
                for (int r = 0; r < 8; ++r) {
                    int rr = row0 + r + (hi << 3);
                    c[r] = bz + w0 * xb[2 * rr] + w1 * xb[2 * rr + 1];
                }
                c = wmma_f16(a0, load_bfrag(w1f, 0, nt, lane), c);
                c = wmma_f16(a1, load_bfrag(w1f, 1, nt, lane), c);
                store_cfrag(gates, row0, nt, lane, c);
            }
        }
        __syncthreads();
        ew_update(gates, c1, h1, tid);      // new h1, c1
        __syncthreads();

        // ---- layer 2: gates = [h1|h2] @ [Wih2;Whh2]^T + b -----------------
        {
            v16h a0 = load_afrag(h1, row0 + m, 0,  hi);
            v16h a1 = load_afrag(h1, row0 + m, 32, hi);
            v16h a2 = load_afrag(h2, row0 + m, 0,  hi);
            v16h a3 = load_afrag(h2, row0 + m, 32, hi);
#pragma unroll
            for (int nt = 0; nt < NT; ++nt) {
                float bz = bias2[nt * 16 + m];
                v8f c;
#pragma unroll
                for (int r = 0; r < 8; ++r) c[r] = bz;
                c = wmma_f16(a0, load_bfrag(w2f, 0, nt, lane), c);
                c = wmma_f16(a1, load_bfrag(w2f, 1, nt, lane), c);
                c = wmma_f16(a2, load_bfrag(w2f, 2, nt, lane), c);
                c = wmma_f16(a3, load_bfrag(w2f, 3, nt, lane), c);
                store_cfrag(gates, row0, nt, lane, c);
            }
        }
        __syncthreads();
        ew_update(gates, c2, h2, tid);      // new h2, c2
        __syncthreads();

        // ---- linear head: out[b][t] = h2 . wlin + b ------------------------
        if (tid < ROWS) {
            float acc = blinS[0];
#pragma unroll
            for (int j = 0; j < HID; ++j) acc += (float)h2[tid * KP + j] * wlin[j];
            obuf[tid * 16 + (t & 15)] = acc;
        }
        // flush 16 timesteps per row as two float4 stores per thread
        if ((t & 15) == 15) {
            __syncthreads();
            int row = tid >> 1, half = tid & 1;
            const float* src = obuf + row * 16 + half * 8;
            float4 v0 = *(const float4*)(src);
            float4 v1 = *(const float4*)(src + 4);
            float* op = out + (size_t)(gbase + row) * TSTEPS + (t - 15) + half * 8;
            *(float4*)(op)     = v0;
            *(float4*)(op + 4) = v1;
        }
        __syncthreads();
    }
}

extern "C" void kernel_launch(void* const* d_in, const int* in_sizes, int n_in,
                              void* d_out, int out_size, void* d_ws, size_t ws_size,
                              hipStream_t stream) {
    (void)in_sizes; (void)n_in; (void)out_size; (void)d_ws; (void)ws_size;
    const float* input = (const float*)d_in[0];
    const float* Wih1  = (const float*)d_in[1];
    const float* Whh1  = (const float*)d_in[2];
    const float* bih1  = (const float*)d_in[3];
    const float* bhh1  = (const float*)d_in[4];
    const float* Wih2  = (const float*)d_in[5];
    const float* Whh2  = (const float*)d_in[6];
    const float* bih2  = (const float*)d_in[7];
    const float* bhh2  = (const float*)d_in[8];
    const float* Wlin  = (const float*)d_in[9];
    const float* blin  = (const float*)d_in[10];
    float* out = (float*)d_out;

    // Raise dynamic-LDS cap (184 KB < 320 KB/WGP). Idempotent & deterministic.
    (void)hipFuncSetAttribute((const void*)lstm2_wmma_kernel,
                              hipFuncAttributeMaxDynamicSharedMemorySize,
                              SMEM_BYTES);

    dim3 grid(BATCH / ROWS);     // 64 persistent blocks
    dim3 block(NTHREADS);        // 4 wave32 each
    lstm2_wmma_kernel<<<grid, block, SMEM_BYTES, stream>>>(
        input, Wih1, Whh1, bih1, bhh1, Wih2, Whh2, bih2, bhh2, Wlin, blin, out);
}
